// Attention_11261404250454
// MI455X (gfx1250) — matrix operand order repaired and checked by reading, not compiled, not run on hardware
//
#include <hip/hip_runtime.h>
#include <cstdint>
#include <cstddef>

// ---------------------------------------------------------------------------
// Types
// ---------------------------------------------------------------------------
typedef __bf16 bf16_t;
typedef __attribute__((ext_vector_type(16))) __bf16 v16bf;
typedef __attribute__((ext_vector_type(8)))  __bf16 v8bf;
typedef __attribute__((ext_vector_type(8)))  float  v8f;

#define WMMA_BF16(a, b, c) \
  __builtin_amdgcn_wmma_f32_16x16x32_bf16(false, (a), false, (b), (short)0, (c), false, false)

constexpr int BATCH  = 2;
constexpr int SEQ    = 2048;
constexpr int DMODEL = 768;
constexpr int NHEAD  = 12;
constexpr int DHEAD  = 64;
constexpr int LD3    = 3 * DMODEL;     // 2304
constexpr int MROWS  = BATCH * SEQ;    // 4096

// ---------------------------------------------------------------------------
// Helpers
// ---------------------------------------------------------------------------
__device__ __forceinline__ bf16_t f2bf(float x) {
  union { float f; unsigned u; } c; c.f = x;
  unsigned r = c.u + 0x7FFFu + ((c.u >> 16) & 1u);   // round-to-nearest-even
  union { unsigned short s; bf16_t b; } o;
  o.s = (unsigned short)(r >> 16);
  return o.b;
}

// A fragment, 16(M) x 32(K) bf16, row-major source with row stride ld.
// Per ISA 7.12.2: lanes 0-15 -> M=lane, K = {0..7, 16..23};
//                 lanes 16-31 -> M=lane-16, K = {8..15, 24..31}.
__device__ __forceinline__ v16bf load_fragA(const bf16_t* __restrict__ base,
                                            int ld, int lane) {
  const int r  = lane & 15;
  const int hf = lane >> 4;
  const bf16_t* p = base + (size_t)r * ld + hf * 8;
  v8bf lo = *reinterpret_cast<const v8bf*>(p);
  v8bf hi = *reinterpret_cast<const v8bf*>(p + 16);
  v16bf f;
#pragma unroll
  for (int i = 0; i < 8; ++i) { f[i] = lo[i]; f[i + 8] = hi[i]; }
  return f;
}

// B fragment, 32(K) x 16(N) bf16. Source stored as [N][K] row-major (row n is
// column n of B), row stride ld. Per sparse-B analog layout:
//   lanes 0-15  -> N=lane,    K = {0..7, 16..23};
//   lanes 16-31 -> N=lane-16, K = {8..15, 24..31}.
__device__ __forceinline__ v16bf load_fragB(const bf16_t* __restrict__ base,
                                            int ld, int lane) {
  const int n = lane & 15;
  const int g = lane >> 4;
  const bf16_t* p = base + (size_t)n * ld + g * 8;
  v8bf lo = *reinterpret_cast<const v8bf*>(p);
  v8bf hi = *reinterpret_cast<const v8bf*>(p + 16);
  v16bf f;
#pragma unroll
  for (int i = 0; i < 8; ++i) { f[i] = lo[i]; f[i + 8] = hi[i]; }
  return f;
}

// ---------------------------------------------------------------------------
// Elementwise prep kernels
// ---------------------------------------------------------------------------
__global__ void cast_f32_to_bf16_kernel(const float* __restrict__ src,
                                        bf16_t* __restrict__ dst, int n) {
  int i = blockIdx.x * blockDim.x + threadIdx.x;
  if (i < n) dst[i] = f2bf(src[i]);
}

// src: fp32 [K][N] row-major  ->  dst: bf16 [N][K] row-major
__global__ void transpose_w_kernel(const float* __restrict__ src,
                                   bf16_t* __restrict__ dst, int K, int N) {
  int i = blockIdx.x * blockDim.x + threadIdx.x;
  if (i >= N * K) return;
  int k = i % K;
  int n = i / K;
  dst[i] = f2bf(src[(size_t)k * N + n]);
}

// qkv bf16 [B,S,3D] -> vT bf16 [B,H,DH,S]
__global__ void transpose_v_kernel(const bf16_t* __restrict__ qkv,
                                   bf16_t* __restrict__ vT) {
  int i = blockIdx.x * blockDim.x + threadIdx.x;
  const int total = BATCH * NHEAD * DHEAD * SEQ;
  if (i >= total) return;
  int s  = i % SEQ;
  int dh = (i / SEQ) % DHEAD;
  int h  = (i / (SEQ * DHEAD)) % NHEAD;
  int b  = i / (SEQ * DHEAD * NHEAD);
  vT[i] = qkv[((size_t)b * SEQ + s) * LD3 + 2 * DMODEL + h * DHEAD + dh];
}

// ---------------------------------------------------------------------------
// GEMM: C[M][N] = A[M][K](bf16) * Wt[N][K]^T(bf16) + bias[N](f32)
// Block = 128 threads = 4 waves; wave w computes a 16(M) x 64(N) strip.
// ---------------------------------------------------------------------------
template <typename OutT>
__global__ void __launch_bounds__(128)
gemm_bf16_kernel(const bf16_t* __restrict__ A, const bf16_t* __restrict__ Wt,
                 const float* __restrict__ bias, OutT* __restrict__ C,
                 int N, int K) {
  const int lane = threadIdx.x & 31;
  const int wave = threadIdx.x >> 5;
  const int m0 = blockIdx.x * 16;
  const int n0 = blockIdx.y * 256 + wave * 64;

  v8f acc[4] = {};
  const bf16_t* aptr = A + (size_t)m0 * K;

  for (int k0 = 0; k0 < K; k0 += 32) {
    __builtin_prefetch(aptr + k0 + 32, 0, 1);
    v16bf a = load_fragA(aptr + k0, K, lane);
#pragma unroll
    for (int t = 0; t < 4; ++t) {
      v16bf b = load_fragB(Wt + (size_t)(n0 + t * 16) * K + k0, K, lane);
      acc[t] = WMMA_BF16(a, b, acc[t]);
    }
  }

  const int hf = lane >> 4;
  const int lc = lane & 15;
#pragma unroll
  for (int t = 0; t < 4; ++t) {
    const int col = n0 + t * 16 + lc;
    const float bs = bias[col];
#pragma unroll
    for (int j = 0; j < 8; ++j) {
      const int row = m0 + j + 8 * hf;
      float v = acc[t][j] + bs;
      if constexpr (sizeof(OutT) == 2) {
        C[(size_t)row * N + col] = f2bf(v);
      } else {
        C[(size_t)row * N + col] = v;
      }
    }
  }
}

// ---------------------------------------------------------------------------
// Flash attention. Block = 128 threads = 4 waves; each wave owns a 16-row Q
// tile for head (b,h) and streams KV in 32-wide chunks with online softmax.
// qkv: bf16 [B,S,3D]; vT: bf16 [B,H,DH,S]; out: bf16 [B,S,D]
// ---------------------------------------------------------------------------
__global__ void __launch_bounds__(128)
flash_attn_kernel(const bf16_t* __restrict__ qkv,
                  const bf16_t* __restrict__ vT,
                  bf16_t* __restrict__ out) {
  const int lane = threadIdx.x & 31;
  const int wave = threadIdx.x >> 5;
  const int hf = lane >> 4;
  const int lc = lane & 15;
  const int bh = blockIdx.y;
  const int b = bh / NHEAD;
  const int h = bh - b * NHEAD;
  const int q0 = (blockIdx.x * 4 + wave) * 16;

  __shared__ __align__(16) bf16_t pbuf[4][16 * 32];

  // Q fragments (A-layout), DH=64 split into two K=32 chunks. Loaded once.
  const bf16_t* qbase = qkv + ((size_t)b * SEQ + q0) * LD3 + h * DHEAD;
  const v16bf qa0 = load_fragA(qbase, LD3, lane);
  const v16bf qa1 = load_fragA(qbase + 32, LD3, lane);

  v8f o[4] = {};
  float m_i[8], l_i[8];
#pragma unroll
  for (int j = 0; j < 8; ++j) { m_i[j] = -1e30f; l_i[j] = 0.0f; }

  const float scale = 0.125f;  // 1/sqrt(64)

  for (int kv0 = 0; kv0 < q0 + 16; kv0 += 32) {
    // ---- scores: S = Q * K^T, two 16x16 N-subtiles over this 32-kv chunk
    v8f s0 = {}, s1 = {};
    {
      const bf16_t* kb0 = qkv + ((size_t)b * SEQ + kv0) * LD3 + DMODEL + h * DHEAD;
      v16bf kf0 = load_fragB(kb0, LD3, lane);
      v16bf kf1 = load_fragB(kb0 + 32, LD3, lane);
      s0 = WMMA_BF16(qa0, kf0, s0);
      s0 = WMMA_BF16(qa1, kf1, s0);
      const bf16_t* kb1 = kb0 + (size_t)16 * LD3;
      v16bf kf2 = load_fragB(kb1, LD3, lane);
      v16bf kf3 = load_fragB(kb1 + 32, LD3, lane);
      s1 = WMMA_BF16(qa0, kf2, s1);
      s1 = WMMA_BF16(qa1, kf3, s1);
    }

    // ---- mask + online softmax (row r = q0 + j + 8*hf, cols striped on lanes)
    float p0v[8], p1v[8];
#pragma unroll
    for (int j = 0; j < 8; ++j) {
      const int row = q0 + j + 8 * hf;
      float v0 = s0[j] * scale;
      float v1 = s1[j] * scale;
      v0 = (kv0 + lc > row) ? -1e30f : v0;
      v1 = (kv0 + 16 + lc > row) ? -1e30f : v1;
      float mj = fmaxf(v0, v1);
#pragma unroll
      for (int off = 1; off < 16; off <<= 1)
        mj = fmaxf(mj, __shfl_xor(mj, off, 32));
      const float m_new = fmaxf(m_i[j], mj);
      const float alpha = __expf(m_i[j] - m_new);
      const float p0 = __expf(v0 - m_new);
      const float p1 = __expf(v1 - m_new);
      float rs = p0 + p1;
#pragma unroll
      for (int off = 1; off < 16; off <<= 1)
        rs += __shfl_xor(rs, off, 32);
      l_i[j] = l_i[j] * alpha + rs;
      m_i[j] = m_new;
      p0v[j] = p0;
      p1v[j] = p1;
      o[0][j] = o[0][j] * alpha;
      o[1][j] = o[1][j] * alpha;
      o[2][j] = o[2][j] * alpha;
      o[3][j] = o[3][j] * alpha;
    }

    // ---- P (C-layout) -> LDS 16x32 row-major -> reread as A-fragment
#pragma unroll
    for (int j = 0; j < 8; ++j) {
      pbuf[wave][(j + 8 * hf) * 32 + lc]      = f2bf(p0v[j]);
      pbuf[wave][(j + 8 * hf) * 32 + 16 + lc] = f2bf(p1v[j]);
    }
    asm volatile("s_wait_dscnt 0x0" ::: "memory");
    v16bf pa;
    {
      const bf16_t* pp = &pbuf[wave][lc * 32 + hf * 8];
      v8bf lo = *reinterpret_cast<const v8bf*>(pp);
      v8bf hi = *reinterpret_cast<const v8bf*>(pp + 16);
#pragma unroll
      for (int i = 0; i < 8; ++i) { pa[i] = lo[i]; pa[i + 8] = hi[i]; }
    }

    // ---- O += P * V  (V pre-transposed: vT[bh][dh][s] -> contiguous kv runs)
#pragma unroll
    for (int n = 0; n < 4; ++n) {
      const bf16_t* vb = vT + ((size_t)bh * DHEAD + n * 16) * SEQ + kv0;
      v16bf vf = load_fragB(vb, SEQ, lane);
      o[n] = WMMA_BF16(pa, vf, o[n]);
    }
  }

  // ---- epilogue: normalize and store bf16 [B,S,D]
#pragma unroll
  for (int j = 0; j < 8; ++j) {
    const float inv = 1.0f / l_i[j];
    const int row = q0 + j + 8 * hf;
#pragma unroll
    for (int n = 0; n < 4; ++n) {
      const int col = h * DHEAD + n * 16 + lc;
      out[((size_t)b * SEQ + row) * DMODEL + col] = f2bf(o[n][j] * inv);
    }
  }
}

// ---------------------------------------------------------------------------
// Launcher
// ---------------------------------------------------------------------------
extern "C" void kernel_launch(void* const* d_in, const int* in_sizes, int n_in,
                              void* d_out, int out_size, void* d_ws, size_t ws_size,
                              hipStream_t stream) {
  (void)in_sizes; (void)n_in; (void)out_size; (void)ws_size;

  const float* x        = (const float*)d_in[0];
  // d_in[1] = attn_mask (causal; implicit in the kernel)
  const float* c_attn_w = (const float*)d_in[2];
  const float* c_attn_b = (const float*)d_in[3];
  const float* c_proj_w = (const float*)d_in[4];
  const float* c_proj_b = (const float*)d_in[5];

  char* ws = (char*)d_ws;
  bf16_t* x_bf   = (bf16_t*)(ws + 0);          //  6,291,456 B  (4096 x 768)
  bf16_t* wT_a   = (bf16_t*)(ws + 6291456);    //  3,538,944 B  (2304 x 768)
  bf16_t* wT_p   = (bf16_t*)(ws + 9830400);    //  1,179,648 B  ( 768 x 768)
  bf16_t* qkv    = (bf16_t*)(ws + 11010048);   // 18,874,368 B  (4096 x 2304)
  bf16_t* vT     = (bf16_t*)(ws + 29884416);   //  6,291,456 B  (24 x 64 x 2048)
  bf16_t* attn_o = (bf16_t*)(ws + 36175872);   //  6,291,456 B  (4096 x 768)

  // 1) cast x to bf16
  {
    int n = MROWS * DMODEL;
    cast_f32_to_bf16_kernel<<<(n + 255) / 256, 256, 0, stream>>>(x, x_bf, n);
  }
  // 2) transpose weights to [N][K] bf16
  {
    int n = LD3 * DMODEL;
    transpose_w_kernel<<<(n + 255) / 256, 256, 0, stream>>>(c_attn_w, wT_a, DMODEL, LD3);
  }
  {
    int n = DMODEL * DMODEL;
    transpose_w_kernel<<<(n + 255) / 256, 256, 0, stream>>>(c_proj_w, wT_p, DMODEL, DMODEL);
  }
  // 3) qkv = x @ c_attn_w + b   (bf16 out)
  gemm_bf16_kernel<bf16_t><<<dim3(MROWS / 16, LD3 / 256), 128, 0, stream>>>(
      x_bf, wT_a, c_attn_b, qkv, LD3, DMODEL);
  // 4) V -> [B,H,DH,S]
  {
    int n = BATCH * NHEAD * DHEAD * SEQ;
    transpose_v_kernel<<<(n + 255) / 256, 256, 0, stream>>>(qkv, vT);
  }
  // 5) flash attention -> attn_o bf16 [B,S,D]
  flash_attn_kernel<<<dim3(SEQ / 64, BATCH * NHEAD), 128, 0, stream>>>(qkv, vT, attn_o);
  // 6) out = attn_o @ c_proj_w + b  (fp32 out)
  gemm_bf16_kernel<float><<<dim3(MROWS / 16, DMODEL / 256), 128, 0, stream>>>(
      attn_o, wT_p, c_proj_b, (float*)d_out, DMODEL, DMODEL);
}
